// Loss_51402168599053
// MI455X (gfx1250) — matrix operand order repaired and checked by reading, not compiled, not run on hardware
//
#include <hip/hip_runtime.h>
#include <math.h>

#define N_PAIR 8192
#define NROWS  16384          // 2*N
#define DIM    256
#define TILE   128
#define KSLAB  32             // one bf16 WMMA K-chunk per slab
#define LDSSTB 40             // bf16 row stride: KSLAB + 8 pad (80 B, 16B-mult)

typedef __attribute__((ext_vector_type(8)))  float  v8f;
typedef __attribute__((ext_vector_type(4)))  __bf16 v4bf;
typedef __attribute__((ext_vector_type(8)))  __bf16 v8bf;
typedef __attribute__((ext_vector_type(16))) __bf16 v16bf;

// ---------------------------------------------------------------------------
// Kernel 1: row-normalize [z1;z2] -> U (16384x256 fp32), zero rowsum.
// ---------------------------------------------------------------------------
__global__ __launch_bounds__(256)
void normalize_kernel(const float* __restrict__ z1, const float* __restrict__ z2,
                      float* __restrict__ U, float* __restrict__ rowsum) {
  int row = blockIdx.x;
  int t   = threadIdx.x;
  const float* src = (row < N_PAIR) ? (z1 + (size_t)row * DIM)
                                    : (z2 + (size_t)(row - N_PAIR) * DIM);
  float v  = src[t];
  float ss = v * v;
  #pragma unroll
  for (int m = 16; m >= 1; m >>= 1) ss += __shfl_xor(ss, m, 32);
  __shared__ float sbuf[8];
  int lane = t & 31, wid = t >> 5;
  if (lane == 0) sbuf[wid] = ss;
  __syncthreads();
  if (wid == 0) {
    float x = (lane < 8) ? sbuf[lane] : 0.0f;
    #pragma unroll
    for (int m = 4; m >= 1; m >>= 1) x += __shfl_xor(x, m, 32);
    if (lane == 0) sbuf[0] = x;
  }
  __syncthreads();
  float inv = 1.0f / fmaxf(sqrtf(sbuf[0]), 1e-12f);   // F.normalize eps
  U[(size_t)row * DIM + t] = v * inv;
  if (t == 0) rowsum[row] = 0.0f;
}

// ---------------------------------------------------------------------------
// Fragment loaders (LDS bf16 -> WMMA register layout, ISA 7.12.2 / 7.12.4).
// A 16x32: half-lanes hold K = [8h,8h+8) U [16+8h,16+8h+8)  -> 2x16B loads.
// B 32x16: half-lanes hold K = [8h,8h+8) U [16+8h,16+8h+8)  -> 2x16B loads.
// ---------------------------------------------------------------------------
__device__ __forceinline__ v16bf load_fragA(const __bf16 (*S)[LDSSTB], int row, int half) {
  v8bf p0 = *(const v8bf*)&S[row][half * 8];
  v8bf p1 = *(const v8bf*)&S[row][16 + half * 8];
  return __builtin_shufflevector(p0, p1, 0,1,2,3,4,5,6,7,8,9,10,11,12,13,14,15);
}
__device__ __forceinline__ v16bf load_fragB(const __bf16 (*S)[LDSSTB], int col, int half) {
  v8bf p0 = *(const v8bf*)&S[col][half * 8];
  v8bf p1 = *(const v8bf*)&S[col][16 + half * 8];
  return __builtin_shufflevector(p0, p1, 0,1,2,3,4,5,6,7,8,9,10,11,12,13,14,15);
}
#define WMMA_BF16(A, B, C) \
  __builtin_amdgcn_wmma_f32_16x16x32_bf16(false, (A), false, (B), (short)0, (C), false, false)

// ---------------------------------------------------------------------------
// Kernel 2: symmetric Gram M = U*U^T, 128x128 tiles, upper triangle only.
// Split-precision: U = hi + lo (bf16);  M ~= hi*hi^T + hi*lo^T + lo*hi^T.
// exp(M/tau) reduced on the fly into rowsum[] (rows always; cols for
// off-diagonal tiles via symmetry). bf16 WMMA 16x16x32, f32 accumulate.
// ---------------------------------------------------------------------------
__global__ __launch_bounds__(256)
void gram_exp_rowsum_kernel(const float* __restrict__ U, float* __restrict__ rowsum) {
  const int bi = blockIdx.y, bj = blockIdx.x;
  if (bi > bj) return;                       // upper triangle + diagonal only

  const int t    = threadIdx.x;
  const int lane = t & 31;
  const int w    = t >> 5;                   // 8 waves
  const int wr   = w >> 2;                   // 0..1 : 64-row strip
  const int wc   = w & 3;                    // 0..3 : 32-col strip
  const int row0 = bi * TILE;
  const int col0 = bj * TILE;
  const int half = lane >> 4;
  const int lrow = lane & 15;

  __shared__ __align__(16) __bf16 Ah[TILE][LDSSTB];
  __shared__ __align__(16) __bf16 Al[TILE][LDSSTB];
  __shared__ __align__(16) __bf16 Bh[TILE][LDSSTB];
  __shared__ __align__(16) __bf16 Bl[TILE][LDSSTB];

  v8f acc[4][2];
  #pragma unroll
  for (int ri = 0; ri < 4; ++ri)
    #pragma unroll
    for (int ci = 0; ci < 2; ++ci) acc[ri][ci] = (v8f){};

  for (int k0 = 0; k0 < DIM; k0 += KSLAB) {
    __syncthreads();
    // Stage 128x32 fp32 slabs of i-rows and j-rows, split into bf16 hi+lo.
    #pragma unroll
    for (int idx = t; idx < TILE * (KSLAB / 4); idx += 256) {
      int r  = idx >> 3;                     // 8 float4 per row
      int c4 = (idx & 7) << 2;
      float4 va = *(const float4*)&U[(size_t)(row0 + r) * DIM + k0 + c4];
      float4 vb = *(const float4*)&U[(size_t)(col0 + r) * DIM + k0 + c4];
      v4bf ah = {(__bf16)va.x, (__bf16)va.y, (__bf16)va.z, (__bf16)va.w};
      v4bf bh = {(__bf16)vb.x, (__bf16)vb.y, (__bf16)vb.z, (__bf16)vb.w};
      v4bf al = {(__bf16)(va.x - (float)ah[0]), (__bf16)(va.y - (float)ah[1]),
                 (__bf16)(va.z - (float)ah[2]), (__bf16)(va.w - (float)ah[3])};
      v4bf bl = {(__bf16)(vb.x - (float)bh[0]), (__bf16)(vb.y - (float)bh[1]),
                 (__bf16)(vb.z - (float)bh[2]), (__bf16)(vb.w - (float)bh[3])};
      *(v4bf*)&Ah[r][c4] = ah;
      *(v4bf*)&Al[r][c4] = al;
      *(v4bf*)&Bh[r][c4] = bh;
      *(v4bf*)&Bl[r][c4] = bl;
    }
    __syncthreads();

    // One K=32 chunk: 3 bf16 WMMAs per 16x16 subtile (hi*hi, hi*lo, lo*hi).
    v16bf bfh[2], bfl[2];
    #pragma unroll
    for (int ci = 0; ci < 2; ++ci) {
      bfh[ci] = load_fragB(Bh, wc * 32 + ci * 16 + lrow, half);
      bfl[ci] = load_fragB(Bl, wc * 32 + ci * 16 + lrow, half);
    }
    #pragma unroll
    for (int ri = 0; ri < 4; ++ri) {
      v16bf afh = load_fragA(Ah, wr * 64 + ri * 16 + lrow, half);
      #pragma unroll
      for (int ci = 0; ci < 2; ++ci) {
        acc[ri][ci] = WMMA_BF16(afh, bfh[ci], acc[ri][ci]);   // hi * hi
        acc[ri][ci] = WMMA_BF16(afh, bfl[ci], acc[ri][ci]);   // hi * lo
      }
    }
    #pragma unroll
    for (int ri = 0; ri < 4; ++ri) {
      v16bf afl = load_fragA(Al, wr * 64 + ri * 16 + lrow, half);
      #pragma unroll
      for (int ci = 0; ci < 2; ++ci)
        acc[ri][ci] = WMMA_BF16(afl, bfh[ci], acc[ri][ci]);   // lo * hi
    }
  }

  // exp(s / 0.5) = exp2(s * 2/ln2)
  const float SCALE = 2.885390081777927f;
  #pragma unroll
  for (int ri = 0; ri < 4; ++ri)
    #pragma unroll
    for (int ci = 0; ci < 2; ++ci)
      #pragma unroll
      for (int r = 0; r < 8; ++r)
        acc[ri][ci][r] = exp2f(acc[ri][ci][r] * SCALE);

  // Row sums: VGPR r -> row m=r (lanes 0-15) / m=r+8 (lanes 16-31); butterfly
  // over the 16-lane halves (xor masks < 16 stay within each half).
  #pragma unroll
  for (int ri = 0; ri < 4; ++ri) {
    int base = row0 + wr * 64 + ri * 16;
    #pragma unroll
    for (int r = 0; r < 8; ++r) {
      float v = acc[ri][0][r] + acc[ri][1][r];
      v += __shfl_xor(v, 1, 32);
      v += __shfl_xor(v, 2, 32);
      v += __shfl_xor(v, 4, 32);
      v += __shfl_xor(v, 8, 32);
      if (lane == 0)  atomicAdd(&rowsum[base + r], v);
      if (lane == 16) atomicAdd(&rowsum[base + r + 8], v);
    }
  }

  // Column sums for off-diagonal tiles (mirror rows of the lower triangle).
  if (bi != bj) {
    #pragma unroll
    for (int ci = 0; ci < 2; ++ci) {
      float v = 0.0f;
      #pragma unroll
      for (int ri = 0; ri < 4; ++ri)
        #pragma unroll
        for (int r = 0; r < 8; ++r) v += acc[ri][ci][r];
      v += __shfl_xor(v, 16, 32);            // fold upper-half rows in
      if (lane < 16)
        atomicAdd(&rowsum[col0 + wc * 32 + ci * 16 + lane], v);
    }
  }
}

// ---------------------------------------------------------------------------
// Kernel 3: positive-pair similarities possim[i] = an_i . bn_i  (exact fp32).
// ---------------------------------------------------------------------------
__global__ __launch_bounds__(256)
void possim_kernel(const float* __restrict__ U, float* __restrict__ possim) {
  int i = blockIdx.x, t = threadIdx.x;
  float p = U[(size_t)i * DIM + t] * U[(size_t)(i + N_PAIR) * DIM + t];
  #pragma unroll
  for (int m = 16; m >= 1; m >>= 1) p += __shfl_xor(p, m, 32);
  __shared__ float sbuf[8];
  int lane = t & 31, wid = t >> 5;
  if (lane == 0) sbuf[wid] = p;
  __syncthreads();
  if (t == 0) {
    float x = 0.0f;
    #pragma unroll
    for (int q = 0; q < 8; ++q) x += sbuf[q];
    possim[i] = x;
  }
}

// ---------------------------------------------------------------------------
// Kernel 4: out = (Sum_i log(rowsum_i - e^2) - 4 * Sum_i possim_i) / 16384
//           (mean flag; *0.5 for sum mode). e^{1/tau}=e^2 is the self term.
// ---------------------------------------------------------------------------
__global__ __launch_bounds__(256)
void finalize_kernel(const float* __restrict__ rowsum, const float* __restrict__ possim,
                     const int* __restrict__ meanflag, float* __restrict__ out) {
  int t = threadIdx.x;
  const float E2 = 7.38905609893065f;
  float slog = 0.0f, spos = 0.0f;
  for (int i = t; i < NROWS; i += 256)  slog += logf(rowsum[i] - E2);
  for (int i = t; i < N_PAIR; i += 256) spos += possim[i];
  #pragma unroll
  for (int m = 16; m >= 1; m >>= 1) {
    slog += __shfl_xor(slog, m, 32);
    spos += __shfl_xor(spos, m, 32);
  }
  __shared__ float sa[8], sb[8];
  int lane = t & 31, wid = t >> 5;
  if (lane == 0) { sa[wid] = slog; sb[wid] = spos; }
  __syncthreads();
  if (t == 0) {
    float SL = 0.0f, SP = 0.0f;
    #pragma unroll
    for (int q = 0; q < 8; ++q) { SL += sa[q]; SP += sb[q]; }
    float total = SL - 4.0f * SP;   // sim/tau = 2*sim; summed twice over pairs
    out[0] = meanflag[0] ? (total / (float)NROWS) : (total * 0.5f);
  }
}

// ---------------------------------------------------------------------------
extern "C" void kernel_launch(void* const* d_in, const int* in_sizes, int n_in,
                              void* d_out, int out_size, void* d_ws, size_t ws_size,
                              hipStream_t stream) {
  const float* z1       = (const float*)d_in[0];
  const float* z2       = (const float*)d_in[1];
  const int*   meanflag = (const int*)d_in[2];
  float* U      = (float*)d_ws;                       // 16384*256 f32 = 16 MB
  float* rowsum = U + (size_t)NROWS * DIM;            // 16384 f32
  float* possim = rowsum + NROWS;                     // 8192 f32
  float* out    = (float*)d_out;

  normalize_kernel<<<NROWS, 256, 0, stream>>>(z1, z2, U, rowsum);
  dim3 grid(NROWS / TILE, NROWS / TILE);
  gram_exp_rowsum_kernel<<<grid, 256, 0, stream>>>(U, rowsum);
  possim_kernel<<<N_PAIR, 256, 0, stream>>>(U, possim);
  finalize_kernel<<<1, 256, 0, stream>>>(rowsum, possim, meanflag, out);
}
